// FullAtomEncoderLayer_32547262169804
// MI455X (gfx1250) — matrix operand order repaired and checked by reading, not compiled, not run on hardware
//
#include <hip/hip_runtime.h>
#include <hip/hip_bf16.h>
#include <stdint.h>

#define B_    2
#define N_    1024
#define K_    32
#define DIM_  384
#define PDIM_ 128
#define MSG_  128
#define H_    16
#define HD_   64
#define KD_   64
#define HID_  64
#define EDIM_ 64
#define A_    14336
#define KA_   48
#define NF_   (B_ * N_)       // 2048
#define EE_   (B_ * N_ * K_)  // 65536
#define ECA_  (B_ * N_ * KA_) // 98304
#define BA_   (B_ * A_)       // 28672

typedef __attribute__((ext_vector_type(16))) __bf16 v16bf;
typedef __attribute__((ext_vector_type(8)))  __bf16 v8bf;
typedef __attribute__((ext_vector_type(8)))  float  v8f;
typedef unsigned short ushort_t;

__device__ __forceinline__ ushort_t bfbits(float f) {
  uint32_t u = __builtin_bit_cast(uint32_t, f);
  return (ushort_t)((u + 0x7FFFu + ((u >> 16) & 1u)) >> 16);  // RNE
}
__device__ __forceinline__ v8bf ld8(const ushort_t* p) {
  return *reinterpret_cast<const v8bf*>(p);  // 16B global_load_b128
}
__device__ __forceinline__ v16bf mk16(v8bf lo, v8bf hi) {
  return __builtin_shufflevector(lo, hi, 0,1,2,3,4,5,6,7,8,9,10,11,12,13,14,15);
}
__device__ __forceinline__ v8f wmma_bf(v16bf a, v16bf b, v8f c) {
  return __builtin_amdgcn_wmma_f32_16x16x32_bf16(false, a, false, b,
                                                 (short)0, c, false, false);
}

// ---------------------------------------------------------------------------
// Shared epilogue: v = act(v + bias (+Cin)) [*rowMask]; store f32 and/or bf16
// ---------------------------------------------------------------------------
__device__ __forceinline__ void epilogue16(
    v8f acc, int tm, int tn, int hf, int l16,
    float* __restrict__ C, int ldc, ushort_t* __restrict__ Cb, int ldcb,
    const float* __restrict__ bias, const float* __restrict__ Cin, int ldcin,
    const unsigned char* __restrict__ rowMask, int act)
{
  const int col = tn + l16;
  const float bv = bias ? bias[col] : 0.0f;
#pragma unroll
  for (int r = 0; r < 8; ++r) {
    const int row = tm + hf * 8 + r;
    float v = acc[r] + bv;
    if (Cin) v += Cin[(long)row * ldcin + col];
    if (act == 1)      v = 0.5f * v * (1.0f + erff(v * 0.70710678118f));
    else if (act == 2) v = (v > 0.0f) ? v : 0.01f * v;
    if (rowMask && !rowMask[row]) v = 0.0f;
    if (C)  C[(long)row * ldc + col] = v;
    if (Cb) Cb[(long)row * ldcb + col] = bfbits(v);
  }
}

// ---------------------------------------------------------------------------
// Main GEMM: 32x32 tile per wave = 2x2 WMMA tiles, 4 accumulators.
// Per K-step: 8x global_load_b128 -> 4x v_wmma (2 loads/WMMA, fragments
// land directly in the wave32 WMMA layout; zero conversion VALU in loop).
// ---------------------------------------------------------------------------
template <int KD>
__global__ void gemm2_bf16_kernel(
    const ushort_t* __restrict__ A, int lda, long sA,
    const ushort_t* __restrict__ Bt, int ldb, long sB,
    float* __restrict__ C, int ldc, long sC,
    ushort_t* __restrict__ Cb, int ldcb, long sCb,
    const float* __restrict__ bias,
    const float* __restrict__ Cin, int ldcin, long sCin,
    const unsigned char* __restrict__ rowMask,
    int M, int Nn, int act)
{
  const int batch = blockIdx.y;
  A  += (long)batch * sA;
  Bt += (long)batch * sB;
  if (C)   C   += (long)batch * sC;
  if (Cb)  Cb  += (long)batch * sCb;
  if (Cin) Cin += (long)batch * sCin;

  const int tilesN = Nn >> 5;
  const int tile   = blockIdx.x * (blockDim.x >> 5) + (threadIdx.x >> 5);
  if (tile >= (M >> 5) * tilesN) return;  // whole-wave exit: EXEC all-ones
  const int tm   = (tile / tilesN) << 5;
  const int tn   = (tile % tilesN) << 5;
  const int lane = threadIdx.x & 31;
  const int hf   = lane >> 4;
  const int l16  = lane & 15;

  const ushort_t* __restrict__ Ap0 = A + (long)(tm + l16) * lda + hf * 8;
  const ushort_t* __restrict__ Ap1 = Ap0 + (long)16 * lda;
  const ushort_t* __restrict__ Bp0 = Bt + (long)(tn + l16) * ldb + hf * 8;
  const ushort_t* __restrict__ Bp1 = Bp0 + (long)16 * ldb;

  v8f acc00 = {}, acc01 = {}, acc10 = {}, acc11 = {};
#pragma unroll 2
  for (int s = 0; s < KD / 32; ++s) {
    const int k0 = s * 32;
    v16bf a0 = mk16(ld8(Ap0 + k0), ld8(Ap0 + k0 + 16));
    v16bf a1 = mk16(ld8(Ap1 + k0), ld8(Ap1 + k0 + 16));
    v16bf b0 = mk16(ld8(Bp0 + k0), ld8(Bp0 + k0 + 16));
    v16bf b1 = mk16(ld8(Bp1 + k0), ld8(Bp1 + k0 + 16));
    acc00 = wmma_bf(a0, b0, acc00);
    acc01 = wmma_bf(a0, b1, acc01);
    acc10 = wmma_bf(a1, b0, acc10);
    acc11 = wmma_bf(a1, b1, acc11);
  }
  epilogue16(acc00, tm,      tn,      hf, l16, C, ldc, Cb, ldcb, bias, Cin, ldcin, rowMask, act);
  epilogue16(acc01, tm,      tn + 16, hf, l16, C, ldc, Cb, ldcb, bias, Cin, ldcin, rowMask, act);
  epilogue16(acc10, tm + 16, tn,      hf, l16, C, ldc, Cb, ldcb, bias, Cin, ldcin, rowMask, act);
  epilogue16(acc11, tm + 16, tn + 16, hf, l16, C, ldc, Cb, ldcb, bias, Cin, ldcin, rowMask, act);
}

// Fallback 16x16-tile GEMM (only for N=16: the attn-bias projection, K=128)
template <int KD>
__global__ void gemm1_bf16_kernel(
    const ushort_t* __restrict__ A, int lda, long sA,
    const ushort_t* __restrict__ Bt, int ldb, long sB,
    float* __restrict__ C, int ldc, long sC,
    ushort_t* __restrict__ Cb, int ldcb, long sCb,
    const float* __restrict__ bias,
    const float* __restrict__ Cin, int ldcin, long sCin,
    const unsigned char* __restrict__ rowMask,
    int M, int Nn, int act)
{
  const int batch = blockIdx.y;
  A  += (long)batch * sA;
  Bt += (long)batch * sB;
  if (C)   C   += (long)batch * sC;
  if (Cb)  Cb  += (long)batch * sCb;
  if (Cin) Cin += (long)batch * sCin;

  const int tilesN = Nn >> 4;
  const int tile   = blockIdx.x * (blockDim.x >> 5) + (threadIdx.x >> 5);
  if (tile >= (M >> 4) * tilesN) return;
  const int tm   = (tile / tilesN) << 4;
  const int tn   = (tile % tilesN) << 4;
  const int lane = threadIdx.x & 31;
  const int hf   = lane >> 4;
  const int l16  = lane & 15;

  const ushort_t* __restrict__ Ap = A + (long)(tm + l16) * lda + hf * 8;
  const ushort_t* __restrict__ Bp = Bt + (long)(tn + l16) * ldb + hf * 8;

  v8f acc = {};
#pragma unroll 4
  for (int s = 0; s < KD / 32; ++s) {
    const int k0 = s * 32;
    v16bf a = mk16(ld8(Ap + k0), ld8(Ap + k0 + 16));
    v16bf b = mk16(ld8(Bp + k0), ld8(Bp + k0 + 16));
    acc = wmma_bf(a, b, acc);
  }
  epilogue16(acc, tm, tn, hf, l16, C, ldc, Cb, ldcb, bias, Cin, ldcin, rowMask, act);
}

// --------------------------- elementwise helpers ---------------------------
__global__ void tobf_kernel(const float* __restrict__ x, ushort_t* __restrict__ y, long n) {
  const long i = (long)blockIdx.x * blockDim.x + threadIdx.x;
  if (i < n) y[i] = bfbits(x[i]);
}

// W row-major [K,N] f32 -> Wt row-major [N,K] bf16
__global__ void transpose_bf16_kernel(const float* __restrict__ W,
                                      ushort_t* __restrict__ Wt, int K, int N) {
  const int i = blockIdx.x * blockDim.x + threadIdx.x;
  if (i < K * N) {
    const int n = i / K, k = i % K;
    Wt[i] = bfbits(W[(long)k * N + n]);
  }
}

// out[row,:] = LN(base[row,:] + srcp[gather,:] + tgtp[node,:]) -> bf16
__global__ void assemble_ln_kernel(
    const float* __restrict__ base, const float* __restrict__ srcp,
    const float* __restrict__ tgtp, const int* __restrict__ eidx,
    const float* __restrict__ g, const float* __restrict__ bb,
    ushort_t* __restrict__ out)
{
  const int row  = blockIdx.x;
  const int t    = threadIdx.x;  // 128
  const int node = row / K_;
  const int bat  = node / N_;
  const int src  = bat * N_ + eidx[row];
  float x = base[(long)row * 128 + t] + srcp[(long)src * 128 + t]
          + tgtp[(long)node * 128 + t];
  __shared__ float red[128];
  red[t] = x; __syncthreads();
  for (int o = 64; o > 0; o >>= 1) { if (t < o) red[t] += red[t + o]; __syncthreads(); }
  const float mean = red[0] * (1.0f / 128.0f);
  __syncthreads();
  const float d = x - mean;
  red[t] = d * d; __syncthreads();
  for (int o = 64; o > 0; o >>= 1) { if (t < o) red[t] += red[t + o]; __syncthreads(); }
  const float var = red[0] * (1.0f / 128.0f);
  out[(long)row * 128 + t] = bfbits(d * rsqrtf(var + 1e-5f) * g[t] + bb[t]);
}

// EGAT: one wave32 per 48-edge centroid segment; softmax; agg -> bf16
__global__ void egat_kernel(
    const float* __restrict__ q, const float* __restrict__ kk,
    const float* __restrict__ v, const float* __restrict__ cfeat,
    const float* __restrict__ We, const float* __restrict__ wd,
    const float* __restrict__ cpos, const float* __restrict__ apos,
    const int* __restrict__ ki, ushort_t* __restrict__ agg)
{
  const int w    = threadIdx.x >> 5;
  const int lane = threadIdx.x & 31;
  const int seg  = blockIdx.x * 2 + w;
  __shared__ float attn_s[2][KA_];
  const float* qrow = q + (long)seg * 64;
  const float cx = cpos[seg * 3 + 0], cy = cpos[seg * 3 + 1], cz = cpos[seg * 3 + 2];
  const float wdv = wd[0];

  auto logit = [&](int e) -> float {
    const long eg = (long)seg * KA_ + e;
    const int ai = ki[eg];
    float dot = 0.0f, bias = 0.0f;
    for (int d = 0; d < 64; ++d) {
      dot  += qrow[d] * kk[(long)ai * 64 + d];
      bias += cfeat[eg * 64 + d] * We[d];
    }
    const float dx = cx - apos[ai * 3 + 0];
    const float dy = cy - apos[ai * 3 + 1];
    const float dz = cz - apos[ai * 3 + 2];
    return dot * 0.125f + bias - wdv * sqrtf(dx * dx + dy * dy + dz * dz);
  };

  const float l0 = logit(lane);
  const float l1 = (lane < 16) ? logit(32 + lane) : -3.0e38f;
  float m = fmaxf(l0, l1);
  for (int o = 16; o > 0; o >>= 1) m = fmaxf(m, __shfl_xor(m, o, 32));
  const float w0 = __expf(l0 - m);
  const float w1 = (lane < 16) ? __expf(l1 - m) : 0.0f;
  float s = w0 + w1;
  for (int o = 16; o > 0; o >>= 1) s += __shfl_xor(s, o, 32);
  const float inv = 1.0f / (s + 1e-6f);
  attn_s[w][lane] = w0 * inv;
  if (lane < 16) attn_s[w][32 + lane] = w1 * inv;
  __syncthreads();
  const int d0 = lane * 2;
  float a0 = 0.0f, a1 = 0.0f;
  for (int e = 0; e < KA_; ++e) {
    const float a = attn_s[w][e];
    const int ai = ki[(long)seg * KA_ + e];
    a0 += a * v[(long)ai * 64 + d0];
    a1 += a * v[(long)ai * 64 + d0 + 1];
  }
  agg[(long)seg * 64 + d0]     = bfbits(a0);
  agg[(long)seg * 64 + d0 + 1] = bfbits(a1);
}

// og = sigmoid(gl) * maskedMean_k(msg) -> bf16
__global__ void gate_mean_kernel(
    const float* __restrict__ msg, const unsigned char* __restrict__ em,
    const float* __restrict__ gl, ushort_t* __restrict__ og)
{
  const int n = blockIdx.x;
  const int t = threadIdx.x;  // 128
  float s = 0.0f, c = 0.0f;
  for (int k = 0; k < K_; ++k) {
    const float mk = em[n * K_ + k] ? 1.0f : 0.0f;
    s += mk * msg[((long)n * K_ + k) * 128 + t];
    c += mk;
  }
  const float mean = s / (c + 1e-6f);
  const float gate = 1.0f / (1.0f + __expf(-gl[(long)n * 128 + t]));
  og[(long)n * 128 + t] = bfbits(gate * mean);
}

// Per-node masked softmax over K=32 for H=16 heads; amsg[n,h,:]=sum_k aw*msg
__global__ void gat_attn_kernel(
    const float* __restrict__ ab, const unsigned char* __restrict__ em,
    const float* __restrict__ msg, ushort_t* __restrict__ amsg)
{
  const int n = blockIdx.x;
  const int t = threadIdx.x;  // 256
  __shared__ float aw[H_][K_];
  __shared__ float ms[K_][129];
  for (int i = t; i < K_ * 128; i += 256)
    ms[i >> 7][i & 127] = msg[((long)n * K_ + (i >> 7)) * 128 + (i & 127)];
  for (int i = t; i < H_ * K_; i += 256) {
    const int k = i & 31, h = i >> 5;
    float x = ab[((long)n * K_ + k) * H_ + h];
    if (!em[n * K_ + k]) x = -3.0e38f;
    aw[h][k] = x;
  }
  __syncthreads();
  if (t < H_) {
    float m = -3.4e38f;
    for (int k = 0; k < K_; ++k) m = fmaxf(m, aw[t][k]);
    float s = 0.0f;
    for (int k = 0; k < K_; ++k) { const float e = __expf(aw[t][k] - m); aw[t][k] = e; s += e; }
    const float inv = 1.0f / s;
    for (int k = 0; k < K_; ++k) aw[t][k] *= inv;
  }
  __syncthreads();
  for (int i = t; i < H_ * 128; i += 256) {
    const int h = i >> 7, d = i & 127;
    float acc = 0.0f;
    for (int k = 0; k < K_; ++k) acc += aw[h][k] * ms[k][d];
    amsg[((long)n * H_ + h) * 128 + d] = bfbits(acc);
  }
}

__global__ void sig_to_bf_kernel(const float* __restrict__ gl,
                                 const float* __restrict__ x,
                                 ushort_t* __restrict__ y, long n) {
  const long i = (long)blockIdx.x * blockDim.x + threadIdx.x;
  if (i < n) y[i] = bfbits(x[i] / (1.0f + __expf(-gl[i])));
}

__global__ void node_final_kernel(const float* __restrict__ node1,
                                  const float* __restrict__ dh,
                                  const unsigned char* __restrict__ mask,
                                  float* __restrict__ out,
                                  ushort_t* __restrict__ outb, long n) {
  const long i = (long)blockIdx.x * blockDim.x + threadIdx.x;
  if (i < n) {
    const int node = (int)(i / DIM_);
    const float v = mask[node] ? (node1[i] + dh[i]) : 0.0f;
    out[i]  = v;
    outb[i] = bfbits(v);
  }
}

// ---------------------------------------------------------------------------
static inline void launch_gemm(hipStream_t s,
    const ushort_t* A, int lda, long sA,
    const ushort_t* Bt, int ldb, long sB,
    float* C, int ldc, long sC,
    ushort_t* Cb, int ldcb, long sCb,
    const float* bias, const float* Cin, int ldcin, long sCin,
    const unsigned char* rowMask,
    int M, int Nn, int Kd, int act, int batch)
{
  dim3 b(256);
  if ((M & 31) == 0 && (Nn & 31) == 0) {
    const int tiles = (M / 32) * (Nn / 32);
    dim3 g((tiles + 7) / 8, batch, 1);
#define G2(KDV) gemm2_bf16_kernel<KDV><<<g, b, 0, s>>>(A, lda, sA, Bt, ldb, sB, \
    C, ldc, sC, Cb, ldcb, sCb, bias, Cin, ldcin, sCin, rowMask, M, Nn, act)
    switch (Kd) {
      case 64:   G2(64);   break;
      case 128:  G2(128);  break;
      case 256:  G2(256);  break;
      case 384:  G2(384);  break;
      case 1024: G2(1024); break;
      default: break;
    }
#undef G2
  } else {
    const int tiles = (M / 16) * (Nn / 16);
    dim3 g((tiles + 7) / 8, batch, 1);
    if (Kd == 128)
      gemm1_bf16_kernel<128><<<g, b, 0, s>>>(A, lda, sA, Bt, ldb, sB, C, ldc, sC,
                                             Cb, ldcb, sCb, bias, Cin, ldcin, sCin,
                                             rowMask, M, Nn, act);
  }
}

extern "C" void kernel_launch(void* const* d_in, const int* in_sizes, int n_in,
                              void* d_out, int out_size, void* d_ws, size_t ws_size,
                              hipStream_t stream)
{
  (void)in_sizes; (void)n_in; (void)out_size; (void)ws_size;
  const float* node_repr  = (const float*)d_in[0];
  const float* edge_repr  = (const float*)d_in[1];
  const float* atom_feat  = (const float*)d_in[2];
  const float* cfeat      = (const float*)d_in[3];
  const float* cpos       = (const float*)d_in[4];
  const float* apos       = (const float*)d_in[5];
  const float* W_node_src = (const float*)d_in[6];
  const float* W_node_tgt = (const float*)d_in[7];
  const float* W_edge_msg = (const float*)d_in[8];
  const float* msg_ln_g   = (const float*)d_in[9];
  const float* msg_ln_b   = (const float*)d_in[10];
  const float* msg_W1     = (const float*)d_in[11];
  const float* msg_b1     = (const float*)d_in[12];
  const float* msg_W2     = (const float*)d_in[13];
  const float* msg_b2     = (const float*)d_in[14];
  const float* eg_Wq      = (const float*)d_in[15];
  const float* eg_Wk      = (const float*)d_in[16];
  const float* eg_Wv      = (const float*)d_in[17];
  const float* eg_We      = (const float*)d_in[18];
  const float* eg_wd      = (const float*)d_in[19];
  const float* eg_Wo      = (const float*)d_in[20];
  const float* W_gate     = (const float*)d_in[21];
  const float* b_gate     = (const float*)d_in[22];
  const float* W_out      = (const float*)d_in[23];
  const float* W_msg_lin  = (const float*)d_in[24];
  const float* W_attn_b   = (const float*)d_in[25];
  const float* W_gat_v    = (const float*)d_in[26];
  const float* W_gat_gate = (const float*)d_in[27];
  const float* b_gat_gate = (const float*)d_in[28];
  const float* W_gat_out  = (const float*)d_in[29];
  const float* W_out_src  = (const float*)d_in[30];
  const float* W_out_tgt  = (const float*)d_in[31];
  const float* e_ln_g     = (const float*)d_in[32];
  const float* e_ln_b     = (const float*)d_in[33];
  const float* e_W1       = (const float*)d_in[34];
  const float* e_b1       = (const float*)d_in[35];
  const float* e_W2       = (const float*)d_in[36];
  const float* e_b2       = (const float*)d_in[37];
  const int*   edge_index = (const int*)d_in[38];
  const unsigned char* edge_mask = (const unsigned char*)d_in[39];
  const int*   cedge      = (const int*)d_in[40];
  const unsigned char* maskp     = (const unsigned char*)d_in[41];
  const int* ki = cedge + ECA_;

  // ----- f32 arena -----
  float* ws = (float*)d_ws;
  size_t foff = 0;
  auto falloc = [&](size_t n) -> float* { float* p = ws + foff; foff += n; return p; };
  float* node_src = falloc((size_t)NF_ * MSG_);
  float* node_tgt = falloc((size_t)NF_ * MSG_);
  float* qb       = falloc((size_t)NF_ * KD_);
  float* kb       = falloc((size_t)BA_ * KD_);
  float* vb       = falloc((size_t)BA_ * HID_);
  float* bufA     = falloc((size_t)EE_ * MSG_);     // msg0 -> ab
  float* msgb     = falloc((size_t)EE_ * MSG_);
  float* glb      = falloc((size_t)NF_ * MSG_);     // gate logits -> srcp
  float* gl2b     = falloc((size_t)NF_ * H_ * HD_); // gat gate logits -> tgtp
  float* gob      = falloc((size_t)NF_ * H_ * HD_);
  float* dhb      = falloc((size_t)NF_ * DIM_);
  float* node1f   = falloc((size_t)NF_ * DIM_);

  // ----- bf16 arena -----
  ushort_t* bws = (ushort_t*)(ws + foff);
  size_t boff = 0;
  auto balloc = [&](size_t n) -> ushort_t* { ushort_t* p = bws + boff; boff += (n + 7) & ~(size_t)7; return p; };
  ushort_t* nodeR_bf   = balloc((size_t)NF_ * DIM_);
  ushort_t* atom_bf    = balloc((size_t)BA_ * DIM_);
  ushort_t* edge_bf    = balloc((size_t)EE_ * PDIM_);
  ushort_t* mln_bf     = balloc((size_t)EE_ * MSG_);      // msg LN out -> t1
  ushort_t* h1_bf      = balloc((size_t)EE_ * 2 * MSG_);  // shared MLP hidden
  ushort_t* msg_bf     = balloc((size_t)EE_ * MSG_);      // msg -> edge LN out
  ushort_t* agg_bf     = balloc((size_t)NF_ * HID_);
  ushort_t* node1_bf   = balloc((size_t)NF_ * DIM_);
  ushort_t* og_bf      = balloc((size_t)NF_ * MSG_);
  ushort_t* amsg_bf    = balloc((size_t)NF_ * H_ * MSG_);
  ushort_t* gob_bf     = balloc((size_t)NF_ * H_ * HD_);
  ushort_t* nodeout_bf = balloc((size_t)NF_ * DIM_);

  float* node_out = (float*)d_out;
  float* edge_out = (float*)d_out + (size_t)NF_ * DIM_;

  // ----- one-shot converts of kernel inputs -----
  auto tobf = [&](const float* x, ushort_t* y, long n) {
    tobf_kernel<<<(n + 255) / 256, 256, 0, stream>>>(x, y, n);
  };
  tobf(node_repr, nodeR_bf, (long)NF_ * DIM_);
  tobf(atom_feat, atom_bf, (long)BA_ * DIM_);
  tobf(edge_repr, edge_bf, (long)EE_ * PDIM_);

  // ----- weight transposes: [K,N] f32 -> [N,K] bf16 -----
  auto tw = [&](const float* W, int Kd, int Nn) -> ushort_t* {
    ushort_t* p = balloc((size_t)Kd * Nn);
    const int total = Kd * Nn;
    transpose_bf16_kernel<<<(total + 255) / 256, 256, 0, stream>>>(W, p, Kd, Nn);
    return p;
  };
  ushort_t* wt_nsrc  = tw(W_node_src, DIM_, MSG_);
  ushort_t* wt_ntgt  = tw(W_node_tgt, DIM_, MSG_);
  ushort_t* wt_emsg  = tw(W_edge_msg, PDIM_, MSG_);
  ushort_t* wt_mW1   = tw(msg_W1, MSG_, 2 * MSG_);
  ushort_t* wt_mW2   = tw(msg_W2, 2 * MSG_, MSG_);
  ushort_t* wt_q     = tw(eg_Wq, DIM_, KD_);
  ushort_t* wt_k     = tw(eg_Wk, DIM_, KD_);
  ushort_t* wt_v     = tw(eg_Wv, DIM_, HID_);
  ushort_t* wt_o     = tw(eg_Wo, HID_, DIM_);
  ushort_t* wt_gate  = tw(W_gate, DIM_, MSG_);
  ushort_t* wt_out   = tw(W_out, MSG_, DIM_);
  ushort_t* wt_mlin  = tw(W_msg_lin, MSG_, MSG_);
  ushort_t* wt_attn  = tw(W_attn_b, MSG_, H_);
  ushort_t* wt_gv    = tw(W_gat_v, MSG_, H_ * HD_);
  ushort_t* wt_gg    = tw(W_gat_gate, DIM_, H_ * HD_);
  ushort_t* wt_go    = tw(W_gat_out, H_ * HD_, DIM_);
  ushort_t* wt_osrc  = tw(W_out_src, DIM_, PDIM_);
  ushort_t* wt_otgt  = tw(W_out_tgt, DIM_, PDIM_);
  ushort_t* wt_eW1   = tw(e_W1, PDIM_, 2 * PDIM_);
  ushort_t* wt_eW2   = tw(e_W2, 2 * PDIM_, PDIM_);

  // ----- projections -----
  launch_gemm(stream, nodeR_bf, DIM_, 0, wt_nsrc, DIM_, 0, node_src, MSG_, 0,
              nullptr, 0, 0, nullptr, nullptr, 0, 0, nullptr, NF_, MSG_, DIM_, 0, 1);
  launch_gemm(stream, nodeR_bf, DIM_, 0, wt_ntgt, DIM_, 0, node_tgt, MSG_, 0,
              nullptr, 0, 0, nullptr, nullptr, 0, 0, nullptr, NF_, MSG_, DIM_, 0, 1);
  launch_gemm(stream, nodeR_bf, DIM_, 0, wt_q, DIM_, 0, qb, KD_, 0,
              nullptr, 0, 0, nullptr, nullptr, 0, 0, nullptr, NF_, KD_, DIM_, 0, 1);
  launch_gemm(stream, atom_bf, DIM_, 0, wt_k, DIM_, 0, kb, KD_, 0,
              nullptr, 0, 0, nullptr, nullptr, 0, 0, nullptr, BA_, KD_, DIM_, 0, 1);
  launch_gemm(stream, atom_bf, DIM_, 0, wt_v, DIM_, 0, vb, HID_, 0,
              nullptr, 0, 0, nullptr, nullptr, 0, 0, nullptr, BA_, HID_, DIM_, 0, 1);

  // ----- edge message: proj + gather + LN + MLP (mask in epilogue) -----
  launch_gemm(stream, edge_bf, PDIM_, 0, wt_emsg, PDIM_, 0, bufA, MSG_, 0,
              nullptr, 0, 0, nullptr, nullptr, 0, 0, nullptr, EE_, MSG_, PDIM_, 0, 1);
  assemble_ln_kernel<<<EE_, 128, 0, stream>>>(bufA, node_src, node_tgt, edge_index,
                                              msg_ln_g, msg_ln_b, mln_bf);
  launch_gemm(stream, mln_bf, MSG_, 0, wt_mW1, MSG_, 0, nullptr, 0, 0,
              h1_bf, 2 * MSG_, 0, msg_b1, nullptr, 0, 0, nullptr,
              EE_, 2 * MSG_, MSG_, 1, 1);
  launch_gemm(stream, h1_bf, 2 * MSG_, 0, wt_mW2, 2 * MSG_, 0, msgb, MSG_, 0,
              msg_bf, MSG_, 0, msg_b2, nullptr, 0, 0, edge_mask,
              EE_, MSG_, 2 * MSG_, 0, 1);

  // ----- EGAT atoms->centroids, node update -----
  egat_kernel<<<NF_ / 2, 64, 0, stream>>>(qb, kb, vb, cfeat, eg_We, eg_wd,
                                          cpos, apos, ki, agg_bf);
  launch_gemm(stream, agg_bf, HID_, 0, wt_o, HID_, 0, node1f, DIM_, 0,
              node1_bf, DIM_, 0, nullptr, node_repr, DIM_, 0, nullptr,
              NF_, DIM_, HID_, 0, 1);

  // ----- gated mean branch -----
  launch_gemm(stream, node1_bf, DIM_, 0, wt_gate, DIM_, 0, glb, MSG_, 0,
              nullptr, 0, 0, b_gate, nullptr, 0, 0, nullptr, NF_, MSG_, DIM_, 0, 1);
  gate_mean_kernel<<<NF_, 128, 0, stream>>>(msgb, edge_mask, glb, og_bf);
  launch_gemm(stream, og_bf, MSG_, 0, wt_out, MSG_, 0, dhb, DIM_, 0,
              nullptr, 0, 0, nullptr, nullptr, 0, 0, nullptr, NF_, DIM_, MSG_, 0, 1);

  // ----- GAT branch -----
  launch_gemm(stream, msg_bf, MSG_, 0, wt_mlin, MSG_, 0, nullptr, 0, 0,
              mln_bf, MSG_, 0, nullptr, nullptr, 0, 0, nullptr,
              EE_, MSG_, MSG_, 2, 1);                       // leaky-relu -> t1
  launch_gemm(stream, mln_bf, MSG_, 0, wt_attn, MSG_, 0, bufA, H_, 0,
              nullptr, 0, 0, nullptr, nullptr, 0, 0, nullptr, EE_, H_, MSG_, 0, 1);
  gat_attn_kernel<<<NF_, 256, 0, stream>>>(bufA, edge_mask, msgb, amsg_bf);
  // per-head: go[:, h*64:(h+1)*64] = amsg[:, h, :] @ W_gat_v[:, h*64:]
  launch_gemm(stream, amsg_bf, H_ * MSG_, MSG_, wt_gv, MSG_, (long)HD_ * MSG_,
              gob, H_ * HD_, HD_, nullptr, 0, 0, nullptr, nullptr, 0, 0, nullptr,
              NF_, HD_, MSG_, 0, H_);
  launch_gemm(stream, node1_bf, DIM_, 0, wt_gg, DIM_, 0, gl2b, H_ * HD_, 0,
              nullptr, 0, 0, b_gat_gate, nullptr, 0, 0, nullptr,
              NF_, H_ * HD_, DIM_, 0, 1);
  sig_to_bf_kernel<<<((long)NF_ * H_ * HD_ + 255) / 256, 256, 0, stream>>>(
      gl2b, gob, gob_bf, (long)NF_ * H_ * HD_);
  launch_gemm(stream, gob_bf, H_ * HD_, 0, wt_go, H_ * HD_, 0, dhb, DIM_, 0,
              nullptr, 0, 0, nullptr, dhb, DIM_, 0, nullptr, NF_, DIM_, H_ * HD_, 0, 1);
  node_final_kernel<<<((long)NF_ * DIM_ + 255) / 256, 256, 0, stream>>>(
      node1f, dhb, maskp, node_out, nodeout_bf, (long)NF_ * DIM_);

  // ----- edge update -----
  launch_gemm(stream, nodeout_bf, DIM_, 0, wt_osrc, DIM_, 0, glb, PDIM_, 0,
              nullptr, 0, 0, nullptr, nullptr, 0, 0, nullptr, NF_, PDIM_, DIM_, 0, 1);
  launch_gemm(stream, nodeout_bf, DIM_, 0, wt_otgt, DIM_, 0, gl2b, PDIM_, 0,
              nullptr, 0, 0, nullptr, nullptr, 0, 0, nullptr, NF_, PDIM_, DIM_, 0, 1);
  assemble_ln_kernel<<<EE_, 128, 0, stream>>>(edge_repr, glb, gl2b, edge_index,
                                              e_ln_b ? e_ln_g : e_ln_g, e_ln_b, msg_bf);
  launch_gemm(stream, msg_bf, PDIM_, 0, wt_eW1, PDIM_, 0, nullptr, 0, 0,
              h1_bf, 2 * PDIM_, 0, e_b1, nullptr, 0, 0, nullptr,
              EE_, 2 * PDIM_, PDIM_, 1, 1);
  launch_gemm(stream, h1_bf, 2 * PDIM_, 0, wt_eW2, 2 * PDIM_, 0, edge_out, PDIM_, 0,
              nullptr, 0, 0, e_b2, edge_repr, PDIM_, 0, nullptr,
              EE_, PDIM_, 2 * PDIM_, 0, 1);
}
